// EnhancedGeoGNN_44306882625632
// MI455X (gfx1250) — matrix operand order repaired and checked by reading, not compiled, not run on hardware
//
// EnhancedGeoGNN (GATv2, 4 layers, HID=128) for MI455X / gfx1250.
//
// Strategy (see analysis): all dense projections run through
// v_wmma_f32_16x16x32_bf16 (bf16 A/B built in-register from f32 activations,
// weights pre-transposed+pre-converted to bf16 each launch); the edge-wise
// attention softmax/aggregation runs as wave32 wave-per-edge kernels with
// f32 global atomics (the HBM-bound critical path at 23.3 TB/s).
//
// Param pointer order = jax pytree flatten (sorted dict keys):
//   0 x, 1 edge_index, 2 edge_weight,
//   3..12  clf:    W1 W2 W3 b1 b2 b3 be1 be2 g1 g2
//   13..20 feat:   W1 W2 b1 b2 be1 be2 g1 g2
//   21..36 ffn[l]: W1 W2 b1 b2            (l = 0..3)
//   37..44 fusion: W1 W2 b1 b2 be1 be2 g1 g2
//   45..72 gat[l]: We Wl Wr att bias bl br (l = 0..3)
//   73     layer_weights
//   74..81 ln[l]:  b g                    (l = 0..3)
//   82..89 spatial:W1 W2 b1 b2 be1 be2 g1 g2

#include <hip/hip_runtime.h>
#include <hip/hip_bf16.h>

typedef __attribute__((ext_vector_type(16))) __bf16 v16bf;
typedef __attribute__((ext_vector_type(8)))  float  v8f;

union Frag { unsigned int u[8]; v16bf v; };

__device__ __forceinline__ unsigned short f2bf(float f) {
  unsigned int u = __float_as_uint(f);
  u += 0x7fffu + ((u >> 16) & 1u);          // round-to-nearest-even
  return (unsigned short)(u >> 16);
}

__device__ __forceinline__ float gelu_f(float x) {
  return 0.5f * x * (1.0f + erff(x * 0.70710678118654752f));
}

__device__ __forceinline__ float wave_sum(float v) {
  #pragma unroll
  for (int off = 16; off > 0; off >>= 1) v += __shfl_xor(v, off, 32);
  return v;
}

__device__ __forceinline__ void atomicMaxF(float* addr, float val) {
  int* ai = (int*)addr;
  int cur = __float_as_int(*addr);
  while (__int_as_float(cur) < val) {
    int prev = atomicCAS(ai, cur, __float_as_int(val));
    if (prev == cur) break;
    cur = prev;
  }
}

// ---------------------------------------------------------------------------
// WMMA GEMM: C[nrows x ncols] = A[nrows x K](f32) * B[K x ncols](bf16,
// stored transposed as BT[col][K]) (+bias) (+=C). NT = ncols/16 tiles.
// Block = 256 threads = 8 waves; each wave owns 16 rows, all NT col tiles.
// A fragment: lane(hi=l>>4,m=l&15) holds
// row m, K = hi*16 + 0..15 (pairs packed in dwords), matching B.
// B fragment: lane holds col m, K = hi*16 + 0..15 (contiguous in BT row).
// C layout: VGPR r -> row (hi*8 + r), col = m.
// ---------------------------------------------------------------------------
template <int NT>
__global__ __launch_bounds__(256) void k_gemm(
    const float* __restrict__ A, int lda,
    const unsigned short* __restrict__ BT, int ldb,
    const float* __restrict__ bias,
    float* __restrict__ C, int ldc,
    int nrows, int K, int ncols_store, int accum)
{
  const int lane = threadIdx.x & 31;
  const int wave = threadIdx.x >> 5;
  const int hi   = lane >> 4;
  const int m    = lane & 15;
  const int row0 = (blockIdx.x * 8 + wave) * 16;
  if (row0 >= nrows) return;                 // wave-uniform: EXEC stays all-1s
  int rload = row0 + m;
  if (rload >= nrows) rload = nrows - 1;     // clamp; stores are guarded
  const float* arow = A + (size_t)rload * lda;

  v8f acc[NT];
  #pragma unroll
  for (int nt = 0; nt < NT; ++nt)
    #pragma unroll
    for (int q = 0; q < 8; ++q) acc[nt][q] = 0.0f;

  for (int kc = 0; kc < K; kc += 32) {
    Frag af;
    const float* p0 = arow + kc + hi * 16;
    const float* p1 = p0 + 8;
    #pragma unroll
    for (int v = 0; v < 4; ++v) {
      af.u[v]     = (unsigned)f2bf(p0[2 * v]) | ((unsigned)f2bf(p0[2 * v + 1]) << 16);
      af.u[4 + v] = (unsigned)f2bf(p1[2 * v]) | ((unsigned)f2bf(p1[2 * v + 1]) << 16);
    }
    #pragma unroll
    for (int nt = 0; nt < NT; ++nt) {
      const int col = nt * 16 + m;
      const unsigned int* bp =
          (const unsigned int*)(BT + (size_t)col * ldb + kc + hi * 16);
      Frag bf_;
      #pragma unroll
      for (int q = 0; q < 8; ++q) bf_.u[q] = bp[q];
      acc[nt] = __builtin_amdgcn_wmma_f32_16x16x32_bf16(
          false, af.v, false, bf_.v, (short)0, acc[nt], false, false);
    }
  }

  #pragma unroll
  for (int r = 0; r < 8; ++r) {
    const int rr = row0 + hi * 8 + r;
    if (rr < nrows) {
      #pragma unroll
      for (int nt = 0; nt < NT; ++nt) {
        const int col = nt * 16 + m;
        if (col < ncols_store) {
          float v = acc[nt][r];
          if (bias) v += bias[col];
          float* cp = C + (size_t)rr * ldc + col;
          if (accum) *cp += v; else *cp = v;
        }
      }
    }
  }
}

// ---------------------------------------------------------------------------
// Fused epilogue: dst = [gelu]([LN](src + prebias)) + res. Wave per row.
// ---------------------------------------------------------------------------
__global__ __launch_bounds__(256) void k_post(
    const float* __restrict__ src, float* __restrict__ dst,
    const float* __restrict__ res, const float* __restrict__ prebias,
    const float* __restrict__ g, const float* __restrict__ b,
    int nrows, int ncols, int do_ln, int do_gelu)
{
  const int lane = threadIdx.x & 31;
  const int wave = threadIdx.x >> 5;
  const int row  = blockIdx.x * 8 + wave;
  if (row >= nrows) return;
  const int nj = ncols >> 5;                 // 4 (128 cols) or 8 (256 cols)
  const float* sr = src + (size_t)row * ncols;
  float vals[8];
  #pragma unroll
  for (int j = 0; j < 8; ++j)
    if (j < nj) {
      float v = sr[lane + 32 * j];
      if (prebias) v += prebias[lane + 32 * j];
      vals[j] = v;
    }
  if (do_ln) {
    float s = 0.f, s2 = 0.f;
    #pragma unroll
    for (int j = 0; j < 8; ++j)
      if (j < nj) { s += vals[j]; s2 += vals[j] * vals[j]; }
    s = wave_sum(s); s2 = wave_sum(s2);
    const float mu = s / ncols;
    const float var = s2 / ncols - mu * mu;
    const float rstd = rsqrtf(var + 1e-5f);
    #pragma unroll
    for (int j = 0; j < 8; ++j)
      if (j < nj)
        vals[j] = (vals[j] - mu) * rstd * g[lane + 32 * j] + b[lane + 32 * j];
  }
  if (do_gelu) {
    #pragma unroll
    for (int j = 0; j < 8; ++j)
      if (j < nj) vals[j] = gelu_f(vals[j]);
  }
  float* dr = dst + (size_t)row * ncols;
  #pragma unroll
  for (int j = 0; j < 8; ++j)
    if (j < nj) {
      float v = vals[j];
      if (res) v += res[(size_t)row * ncols + lane + 32 * j];
      dr[lane + 32 * j] = v;
    }
}

// ---------------------------------------------------------------------------
// GATv2 edge passes (wave per edge; lane = head-dim, j = head).
// ---------------------------------------------------------------------------
__global__ __launch_bounds__(256) void k_edge1(
    const int* __restrict__ ei, const float* __restrict__ ew,
    const float* __restrict__ xl, const float* __restrict__ xr,
    const float* __restrict__ We, const float* __restrict__ att,
    float* __restrict__ ealpha, float* __restrict__ amax, int E)
{
  const int e = blockIdx.x * 8 + (threadIdx.x >> 5);
  const int lane = threadIdx.x & 31;
  if (e >= E) return;
  const int s = ei[e], t = ei[E + e];
  const float w = ew[e];
  const float* ls = xl + (size_t)s * 128;
  const float* rt = xr + (size_t)t * 128;
  float al[4];
  #pragma unroll
  for (int j = 0; j < 4; ++j) {
    const int d = j * 32 + lane;
    float mm = ls[d] + rt[d] + w * We[d];
    mm = (mm > 0.f) ? mm : 0.2f * mm;        // leaky_relu
    al[j] = wave_sum(mm * att[d]);
  }
  if (lane == 0) {
    #pragma unroll
    for (int j = 0; j < 4; ++j) {
      ealpha[(size_t)e * 4 + j] = al[j];
      atomicMaxF(&amax[(size_t)t * 4 + j], al[j]);
    }
  }
}

__global__ void k_edge2(const int* __restrict__ ei, float* __restrict__ ealpha,
                        const float* __restrict__ amax, float* __restrict__ den,
                        int E)
{
  const int i = blockIdx.x * 256 + threadIdx.x;
  if (i >= E * 4) return;
  const int e = i >> 2, j = i & 3;
  const int t = ei[E + e];
  const float ex = expf(ealpha[i] - amax[(size_t)t * 4 + j]);
  ealpha[i] = ex;
  atomicAdd(&den[(size_t)t * 4 + j], ex);
}

__global__ __launch_bounds__(256) void k_edge3(
    const int* __restrict__ ei, const float* __restrict__ xl,
    const float* __restrict__ ealpha, const float* __restrict__ den,
    float* __restrict__ agg, int E)
{
  const int e = blockIdx.x * 8 + (threadIdx.x >> 5);
  const int lane = threadIdx.x & 31;
  if (e >= E) return;
  const int s = ei[e], t = ei[E + e];
  const float* ls = xl + (size_t)s * 128;
  #pragma unroll
  for (int j = 0; j < 4; ++j) {
    const float a = ealpha[(size_t)e * 4 + j] / (den[(size_t)t * 4 + j] + 1e-16f);
    const int d = j * 32 + lane;
    atomicAdd(&agg[(size_t)t * 128 + d], ls[d] * a);
  }
}

// ---------------------------------------------------------------------------
// Small helpers.
// ---------------------------------------------------------------------------
__global__ void k_convT(const float* __restrict__ W, unsigned short* __restrict__ WT,
                        int fin, int fout, int Kpad, int foutPad)
{
  const int i = blockIdx.x * 256 + threadIdx.x;
  if (i >= foutPad * Kpad) return;
  const int o = i / Kpad, k = i - o * Kpad;
  const float v = (k < fin && o < fout) ? W[(size_t)k * fout + o] : 0.0f;
  WT[i] = f2bf(v);
}

__global__ void k_fill(float* p, float v, int n) {
  const int i = blockIdx.x * 256 + threadIdx.x;
  if (i < n) p[i] = v;
}

__global__ void k_split(const float* __restrict__ x, float* __restrict__ cP,
                        float* __restrict__ fP, int N)
{
  const int i = blockIdx.x * 256 + threadIdx.x;
  if (i >= N * 32) return;
  const int r = i >> 5, c = i & 31;
  const float* xr_ = x + (size_t)r * 16;
  cP[i] = (c < 3)  ? xr_[c]     : 0.0f;
  fP[i] = (c < 13) ? xr_[3 + c] : 0.0f;
}

__global__ void k_softmax4(const float* __restrict__ lw, float* __restrict__ wsm) {
  if (blockIdx.x == 0 && threadIdx.x == 0) {
    float m = fmaxf(fmaxf(lw[0], lw[1]), fmaxf(lw[2], lw[3]));
    float e0 = expf(lw[0] - m), e1 = expf(lw[1] - m);
    float e2 = expf(lw[2] - m), e3 = expf(lw[3] - m);
    const float s = e0 + e1 + e2 + e3;
    wsm[0] = e0 / s; wsm[1] = e1 / s; wsm[2] = e2 / s; wsm[3] = e3 / s;
  }
}

__global__ void k_axpy(float* __restrict__ acc, const float* __restrict__ h,
                       const float* __restrict__ wsm, int li, int n)
{
  const int i = blockIdx.x * 256 + threadIdx.x;
  if (i < n) acc[i] += wsm[li] * h[i];
}

// ---------------------------------------------------------------------------
extern "C" void kernel_launch(void* const* d_in, const int* in_sizes, int n_in,
                              void* d_out, int out_size, void* d_ws, size_t ws_size,
                              hipStream_t stream)
{
  (void)n_in; (void)out_size; (void)ws_size;
  const float* x  = (const float*)d_in[0];
  const int*   ei = (const int*)d_in[1];
  const float* ew = (const float*)d_in[2];
  const int N = in_sizes[0] / 16;
  const int E = in_sizes[2];
  auto F = [&](int i) { return (const float*)d_in[i]; };

  const float *clf_W1 = F(3),  *clf_W2 = F(4),  *clf_W3 = F(5);
  const float *clf_b1 = F(6),  *clf_b2 = F(7),  *clf_b3 = F(8);
  const float *clf_be1 = F(9), *clf_be2 = F(10), *clf_g1 = F(11), *clf_g2 = F(12);
  const float *ft_W1 = F(13), *ft_W2 = F(14), *ft_b1 = F(15), *ft_b2 = F(16);
  const float *ft_be1 = F(17), *ft_be2 = F(18), *ft_g1 = F(19), *ft_g2 = F(20);
  const float *fu_W1 = F(37), *fu_W2 = F(38), *fu_b1 = F(39), *fu_b2 = F(40);
  const float *fu_be1 = F(41), *fu_be2 = F(42), *fu_g1 = F(43), *fu_g2 = F(44);
  const float *lw = F(73);
  const float *sp_W1 = F(82), *sp_W2 = F(83), *sp_b1 = F(84), *sp_b2 = F(85);
  const float *sp_be1 = F(86), *sp_be2 = F(87), *sp_g1 = F(88), *sp_g2 = F(89);

  // ---- workspace carve-out ------------------------------------------------
  char* ws = (char*)d_ws;
  size_t off = 0;
  auto alloc = [&](size_t bytes) -> char* {
    char* p = ws + off;
    off += (bytes + 255) & ~(size_t)255;
    return p;
  };
  unsigned short* spW1T = (unsigned short*)alloc(128 * 32 * 2);
  unsigned short* spW2T = (unsigned short*)alloc(128 * 128 * 2);
  unsigned short* ftW1T = (unsigned short*)alloc(128 * 32 * 2);
  unsigned short* ftW2T = (unsigned short*)alloc(128 * 128 * 2);
  unsigned short* fuW1T = (unsigned short*)alloc(256 * 256 * 2);
  unsigned short* fuW2T = (unsigned short*)alloc(128 * 256 * 2);
  unsigned short *gWlT[4], *gWrT[4], *fW1T[4], *fW2T[4];
  for (int l = 0; l < 4; ++l) {
    gWlT[l] = (unsigned short*)alloc(128 * 128 * 2);
    gWrT[l] = (unsigned short*)alloc(128 * 128 * 2);
    fW1T[l] = (unsigned short*)alloc(256 * 128 * 2);
    fW2T[l] = (unsigned short*)alloc(128 * 256 * 2);
  }
  unsigned short* cW1T = (unsigned short*)alloc(256 * 128 * 2);
  unsigned short* cW2T = (unsigned short*)alloc(128 * 256 * 2);
  unsigned short* cW3T = (unsigned short*)alloc(16 * 128 * 2);

  float* bh    = (float*)alloc((size_t)N * 128 * 4);
  float* bhres = (float*)alloc((size_t)N * 128 * 4);
  float* bxl   = (float*)alloc((size_t)N * 128 * 4);
  float* bxr   = (float*)alloc((size_t)N * 128 * 4);
  float* bagg  = (float*)alloc((size_t)N * 128 * 4);
  float* bhfin = (float*)alloc((size_t)N * 128 * 4);
  float* bt256 = (float*)alloc((size_t)N * 256 * 4);
  float* beal  = (float*)alloc((size_t)E * 4 * 4);
  float* bamax = (float*)alloc((size_t)N * 4 * 4);
  float* bden  = (float*)alloc((size_t)N * 4 * 4);
  float* bwsm  = (float*)alloc(64);
  float* coordsP = beal;                 // early reuse of edge buffer
  float* featsP  = beal + (size_t)N * 32;

  // ---- launch helpers -----------------------------------------------------
  auto cdiv = [](int a, int b) { return (a + b - 1) / b; };
  auto conv = [&](const float* W, unsigned short* WT, int fin, int fout,
                  int Kpad, int foutPad) {
    k_convT<<<cdiv(foutPad * Kpad, 256), 256, 0, stream>>>(W, WT, fin, fout, Kpad, foutPad);
  };
  auto gemm = [&](int NT, const float* A, int lda, const unsigned short* BT, int ldb,
                  const float* bias, float* C, int ldc, int K, int ncs, int accum) {
    dim3 g(cdiv(N, 128));
    if (NT == 16)     k_gemm<16><<<g, 256, 0, stream>>>(A, lda, BT, ldb, bias, C, ldc, N, K, ncs, accum);
    else if (NT == 8) k_gemm<8><<<g, 256, 0, stream>>>(A, lda, BT, ldb, bias, C, ldc, N, K, ncs, accum);
    else              k_gemm<1><<<g, 256, 0, stream>>>(A, lda, BT, ldb, bias, C, ldc, N, K, ncs, accum);
  };
  auto post = [&](const float* src, float* dst, const float* res, const float* pb,
                  const float* g, const float* b, int ncols, int do_ln, int do_gelu) {
    k_post<<<cdiv(N, 8), 256, 0, stream>>>(src, dst, res, pb, g, b, N, ncols, do_ln, do_gelu);
  };
  auto fill = [&](float* p, float v, int n) {
    k_fill<<<cdiv(n, 256), 256, 0, stream>>>(p, v, n);
  };

  // ---- weight prep (bf16, transposed, padded) -----------------------------
  conv(sp_W1, spW1T, 3, 128, 32, 128);
  conv(sp_W2, spW2T, 128, 128, 128, 128);
  conv(ft_W1, ftW1T, 13, 128, 32, 128);
  conv(ft_W2, ftW2T, 128, 128, 128, 128);
  conv(fu_W1, fuW1T, 256, 256, 256, 256);
  conv(fu_W2, fuW2T, 256, 128, 256, 128);
  for (int l = 0; l < 4; ++l) {
    conv(F(45 + 7 * l + 1), gWlT[l], 128, 128, 128, 128);   // Wl
    conv(F(45 + 7 * l + 2), gWrT[l], 128, 128, 128, 128);   // Wr
    conv(F(21 + 4 * l + 0), fW1T[l], 128, 256, 128, 256);   // ffn W1
    conv(F(21 + 4 * l + 1), fW2T[l], 256, 128, 256, 128);   // ffn W2
  }
  conv(clf_W1, cW1T, 128, 256, 128, 256);
  conv(clf_W2, cW2T, 256, 128, 256, 128);
  conv(clf_W3, cW3T, 128, 10, 128, 16);

  k_softmax4<<<1, 32, 0, stream>>>(lw, bwsm);
  k_split<<<cdiv(N * 32, 256), 256, 0, stream>>>(x, coordsP, featsP, N);
  fill(bhfin, 0.0f, N * 128);

  // ---- encoders -----------------------------------------------------------
  gemm(8, coordsP, 32, spW1T, 32, sp_b1, bxl, 128, 32, 128, 0);
  post(bxl, bxl, nullptr, nullptr, sp_g1, sp_be1, 128, 1, 1);
  gemm(8, bxl, 128, spW2T, 128, sp_b2, bxr, 128, 128, 128, 0);   // sp -> bxr
  post(bxr, bxr, nullptr, nullptr, sp_g2, sp_be2, 128, 1, 1);

  gemm(8, featsP, 32, ftW1T, 32, ft_b1, bxl, 128, 32, 128, 0);
  post(bxl, bxl, nullptr, nullptr, ft_g1, ft_be1, 128, 1, 1);
  gemm(8, bxl, 128, ftW2T, 128, ft_b2, bagg, 128, 128, 128, 0);  // ft -> bagg
  post(bagg, bagg, nullptr, nullptr, ft_g2, ft_be2, 128, 1, 0);  // LN only

  // fusion: concat handled as K-split accumulate
  gemm(16, bxr, 128, fuW1T, 256, fu_b1, bt256, 256, 128, 256, 0);
  gemm(16, bagg, 128, fuW1T + 128, 256, nullptr, bt256, 256, 128, 256, 1);
  post(bt256, bt256, nullptr, nullptr, fu_g1, fu_be1, 256, 1, 1);
  gemm(8, bt256, 256, fuW2T, 256, fu_b2, bh, 128, 256, 128, 0);
  post(bh, bh, nullptr, nullptr, fu_g2, fu_be2, 128, 1, 1);

  // ---- GATv2 layers -------------------------------------------------------
  for (int l = 0; l < 4; ++l) {
    const float* g_We   = F(45 + 7 * l + 0);
    const float* g_att  = F(45 + 7 * l + 3);
    const float* g_bias = F(45 + 7 * l + 4);
    const float* g_bl   = F(45 + 7 * l + 5);
    const float* g_br   = F(45 + 7 * l + 6);
    const float* ln_b   = F(74 + 2 * l + 0);
    const float* ln_g   = F(74 + 2 * l + 1);
    const float* ffn_b1 = F(21 + 4 * l + 2);
    const float* ffn_b2 = F(21 + 4 * l + 3);

    hipMemcpyAsync(bhres, bh, (size_t)N * 128 * 4, hipMemcpyDeviceToDevice, stream);
    gemm(8, bh, 128, gWlT[l], 128, g_bl, bxl, 128, 128, 128, 0);
    gemm(8, bh, 128, gWrT[l], 128, g_br, bxr, 128, 128, 128, 0);

    fill(bamax, -1e30f, N * 4);
    fill(bden, 0.0f, N * 4);
    fill(bagg, 0.0f, N * 128);
    k_edge1<<<cdiv(E, 8), 256, 0, stream>>>(ei, ew, bxl, bxr, g_We, g_att, beal, bamax, E);
    k_edge2<<<cdiv(E * 4, 256), 256, 0, stream>>>(ei, beal, bamax, bden, E);
    k_edge3<<<cdiv(E, 8), 256, 0, stream>>>(ei, bxl, beal, bden, bagg, E);

    // h = gelu(ln(agg + gat.bias)) + h_res
    post(bagg, bh, bhres, g_bias, ln_g, ln_b, 128, 1, 1);

    // FFN: h += gelu(h@W1+b1)@W2 + b2
    gemm(16, bh, 128, fW1T[l], 128, ffn_b1, bt256, 256, 128, 256, 0);
    post(bt256, bt256, nullptr, nullptr, nullptr, nullptr, 256, 0, 1);
    gemm(8, bt256, 256, fW2T[l], 256, ffn_b2, bxl, 128, 256, 128, 0);
    post(bxl, bh, bh, nullptr, nullptr, nullptr, 128, 0, 0);     // h = ffn_out + h

    k_axpy<<<cdiv(N * 128, 256), 256, 0, stream>>>(bhfin, bh, bwsm, l, N * 128);
  }

  // ---- classifier ---------------------------------------------------------
  gemm(16, bhfin, 128, cW1T, 128, clf_b1, bt256, 256, 128, 256, 0);
  post(bt256, bt256, nullptr, nullptr, clf_g1, clf_be1, 256, 1, 1);
  gemm(8, bt256, 256, cW2T, 256, clf_b2, bxl, 128, 256, 128, 0);
  post(bxl, bxl, nullptr, nullptr, clf_g2, clf_be2, 128, 1, 1);
  gemm(1, bxl, 128, cW3T, 128, clf_b3, (float*)d_out, 10, 128, 10, 0);
}
